// DualFormer_33947421508264
// MI455X (gfx1250) — matrix operand order repaired and checked by reading, not compiled, not run on hardware
//
#include <hip/hip_runtime.h>
#include <cstdint>
#include <cstddef>

// ---------------------------------------------------------------------------
// DualFormer for MI455X (gfx1250): bf16 WMMA GEMMs + flash attention,
// async global->LDS tile staging (ASYNCcnt), double buffering, DPP softmax.
// B=4, S1=2048, S2=512, E=1024, H=16, D=64, R=32, FF=4096
// ---------------------------------------------------------------------------

typedef __bf16 bf16;
typedef __attribute__((ext_vector_type(16))) __bf16 v16bf;
typedef __attribute__((ext_vector_type(8)))  float  v8f;
typedef __attribute__((ext_vector_type(4)))  int    v4i;

#define E_DIM 1024
#define H_DIM 16
#define D_HEAD 64
#define R_ROPE 32
#define FF_DIM 4096
#define LN_EPS 1e-5f
#define ATTN_SCALE 0.125f   // 64^-0.5

// ---------------------------------------------------------------------------
// CDNA5 async global->LDS copy (GLOBAL_LOAD_ASYNC_TO_LDS_B128, ASYNCcnt).
// ---------------------------------------------------------------------------
#if defined(__has_builtin)
#  if __has_builtin(__builtin_amdgcn_global_load_async_to_lds_b128) && \
      __has_builtin(__builtin_amdgcn_s_wait_asynccnt)
#    define HAVE_ASYNC_LDS 1
#  endif
#  if __has_builtin(__builtin_amdgcn_update_dpp)
#    define HAVE_DPP 1
#  endif
#endif
#ifndef HAVE_ASYNC_LDS
#  define HAVE_ASYNC_LDS 0
#endif
#ifndef HAVE_DPP
#  define HAVE_DPP 0
#endif

#define AS1 __attribute__((address_space(1)))
#define AS3 __attribute__((address_space(3)))

__device__ __forceinline__ void async_copy16B(void* lds_dst, const void* gsrc) {
#if HAVE_ASYNC_LDS
  __builtin_amdgcn_global_load_async_to_lds_b128(
      (AS1 v4i*)gsrc, (AS3 v4i*)lds_dst, 0, 0);
#else
  *(uint4*)lds_dst = *(const uint4*)gsrc;
#endif
}
__device__ __forceinline__ void async_wait() {
#if HAVE_ASYNC_LDS
  __builtin_amdgcn_s_wait_asynccnt(0);
#endif
}

// ---------------------------------------------------------------------------
// DPP16 row_ror rotation within each 16-lane row (VALU, no LDS traffic).
// A ror{1,2,4,8} butterfly performs a full 16-lane commutative reduction.
// ---------------------------------------------------------------------------
template <int ROR>
__device__ __forceinline__ float dpp_row_ror(float x) {
#if HAVE_DPP
  const int i = __builtin_amdgcn_update_dpp(
      0, __float_as_int(x), 0x120 | ROR, 0xF, 0xF, true);
  return __int_as_float(i);
#else
  return __shfl_xor(x, ROR, 32);  // xor butterfly also reduces the 16-group
#endif
}
__device__ __forceinline__ float row16_max(float x) {
  x = fmaxf(x, dpp_row_ror<1>(x));
  x = fmaxf(x, dpp_row_ror<2>(x));
  x = fmaxf(x, dpp_row_ror<4>(x));
  x = fmaxf(x, dpp_row_ror<8>(x));
  return x;
}
__device__ __forceinline__ float row16_sum(float x) {
  x += dpp_row_ror<1>(x);
  x += dpp_row_ror<2>(x);
  x += dpp_row_ror<4>(x);
  x += dpp_row_ror<8>(x);
  return x;
}

// ---------------------------------------------------------------------------
// WMMA helpers (CDNA5 16x16x32 bf16, f32 accumulate)
// ---------------------------------------------------------------------------
__device__ __forceinline__ v8f wmma_bf16(v16bf a, v16bf b, v8f c) {
  return __builtin_amdgcn_wmma_f32_16x16x32_bf16(false, a, false, b,
                                                 (short)0, c, false, false);
}

// A fragment (16x32, 16-bit): lane holds row M; half h supplies
// contiguous K = 16h .. 16h+15 -> two contiguous 16B chunks.
__device__ __forceinline__ v16bf frag_a_lds(const bf16* __restrict__ row, int h) {
  union { v16bf v; uint4 q[2]; } u;
  u.q[0] = *(const uint4*)(row + 16 * h);
  u.q[1] = *(const uint4*)(row + 16 * h + 8);
  return u.v;
}

// B fragment (32x16, 16-bit), LDS stored as [N][K]: lane = column N,
// half h supplies contiguous K = 16h .. 16h+15.
__device__ __forceinline__ v16bf frag_b_lds(const bf16* __restrict__ col, int h) {
  union { v16bf v; uint4 q[2]; } u;
  u.q[0] = *(const uint4*)(col + 16 * h);
  u.q[1] = *(const uint4*)(col + 16 * h + 8);
  return u.v;
}

// ---------------------------------------------------------------------------
// f32 -> bf16 conversion (weights)
// ---------------------------------------------------------------------------
__global__ __launch_bounds__(256) void convert_f32_bf16(
    const float* __restrict__ x, bf16* __restrict__ y, size_t n) {
  size_t i = (size_t)blockIdx.x * blockDim.x + threadIdx.x;
  size_t stride = (size_t)gridDim.x * blockDim.x;
  for (; i < n; i += stride) y[i] = (bf16)x[i];
}

// ---------------------------------------------------------------------------
// LayerNorm: one row per block (256 threads = 8 waves), bf16 output.
// In-place safe for Tin == bf16.
// ---------------------------------------------------------------------------
template <typename Tin>
__global__ __launch_bounds__(256) void ln_kernel(
    const Tin* __restrict__ x, const float* __restrict__ g,
    const float* __restrict__ b, bf16* __restrict__ y, int C) {
  const size_t row = blockIdx.x;
  const Tin* xr = x + row * (size_t)C;
  bf16* yr = y + row * (size_t)C;
  __shared__ float red[16];

  float s = 0.f, s2 = 0.f;
  for (int i = threadIdx.x; i < C; i += blockDim.x) {
    float v = (float)xr[i];
    s += v; s2 += v * v;
  }
#pragma unroll
  for (int off = 16; off > 0; off >>= 1) {
    s  += __shfl_down(s,  off, 32);
    s2 += __shfl_down(s2, off, 32);
  }
  const int wv = threadIdx.x >> 5;
  if ((threadIdx.x & 31) == 0) { red[wv] = s; red[8 + wv] = s2; }
  __syncthreads();
  if (threadIdx.x == 0) {
    float a = 0.f, a2 = 0.f;
#pragma unroll
    for (int i = 0; i < 8; ++i) { a += red[i]; a2 += red[8 + i]; }
    red[0] = a; red[8] = a2;
  }
  __syncthreads();
  const float mean = red[0] / (float)C;
  const float var  = red[8] / (float)C - mean * mean;
  const float rstd = rsqrtf(var + LN_EPS);
  for (int i = threadIdx.x; i < C; i += blockDim.x) {
    float v = ((float)xr[i] - mean) * rstd * g[i] + b[i];
    yr[i] = (bf16)v;
  }
}

// ---------------------------------------------------------------------------
// Partial RoPE (first R=32 dims per head) f32 in -> bf16 out.
// ---------------------------------------------------------------------------
__global__ __launch_bounds__(1024) void rope_kernel(
    const float* __restrict__ pre, const float* __restrict__ cosT,
    const float* __restrict__ sinT, bf16* __restrict__ out, int S) {
  const int token = blockIdx.x;
  const int s = token % S;
  const int d = threadIdx.x & (D_HEAD - 1);
  const size_t base = (size_t)token * E_DIM + (threadIdx.x & ~(D_HEAD - 1));
  float v = pre[base + d];
  if (d < R_ROPE) {
    const float c  = cosT[(size_t)s * R_ROPE + d];
    const float sn = sinT[(size_t)s * R_ROPE + d];
    const float other = (d < R_ROPE / 2) ? -pre[base + d + R_ROPE / 2]
                                         :  pre[base + d - R_ROPE / 2];
    v = v * c + other * sn;
  }
  out[base + d] = (bf16)v;
}

// ---------------------------------------------------------------------------
// GEMM: C[M,N] = A[M,K](bf16) x W[K,N](bf16) + bias (+GELU, +f32 residual).
// Tile 128x64, K-step 32, double-buffered LDS, async staging, ONE barrier
// per K-step (next tile issued into the other buffer before compute).
// ---------------------------------------------------------------------------
template <bool OUT_BF16, bool GELU, bool RESIDUAL>
__global__ __launch_bounds__(256) void gemm_bf16_wmma(
    const bf16* __restrict__ A, const bf16* __restrict__ W,
    const float* __restrict__ bias, const float* __restrict__ res,
    void* __restrict__ outp, int M, int N, int K) {
  constexpr int BM = 128, BN = 64, KT = 32, SAP = KT + 8;  // 80B stride
  __shared__ __align__(16) bf16 sA[2][BM][SAP];
  __shared__ __align__(16) bf16 sB[2][BN][SAP];  // transposed: [n][k]

  const int t = threadIdx.x;
  const int wave = t >> 5, lane = t & 31;
  const int lm = lane & 15, lh = lane >> 4;
  const int tileM = blockIdx.y * BM, tileN = blockIdx.x * BN;

  const int ar = t >> 1, ac = (t & 1) * 16;   // A-tile coords (16 bf16/thread)
  const int bk = t >> 3, bc = (t & 7) * 8;    // W-tile coords (8 bf16/thread)

  auto load_tile = [&](int k0, int buf) {
    const bf16* asrc = A + (size_t)(tileM + ar) * K + k0 + ac;
    async_copy16B(&sA[buf][ar][ac],     asrc);
    async_copy16B(&sA[buf][ar][ac + 8], asrc + 8);
    const bf16* bsrc = W + (size_t)(k0 + bk) * N + tileN + bc;
#pragma unroll
    for (int j = 0; j < 8; ++j) sB[buf][bc + j][bk] = bsrc[j];  // transpose
  };

  v8f acc[4];
#pragma unroll
  for (int i = 0; i < 4; ++i)
#pragma unroll
    for (int r = 0; r < 8; ++r) acc[i][r] = 0.f;

  const int nk = K / KT;
  load_tile(0, 0);
  for (int it = 0; it < nk; ++it) {
    async_wait();
    __syncthreads();  // tile[buf] visible; all waves done reading buf^1
    if (it + 1 < nk) load_tile((it + 1) * KT, (it + 1) & 1);
    const int buf = it & 1;
    const v16bf af = frag_a_lds(&sA[buf][wave * 16 + lm][0], lh);
#pragma unroll
    for (int tn = 0; tn < 4; ++tn) {
      const v16bf bfrag = frag_b_lds(&sB[buf][tn * 16 + lm][0], lh);
      acc[tn] = wmma_bf16(af, bfrag, acc[tn]);
    }
  }

  // Epilogue. C layout: row = 8*half + r, col = lane&15.
#pragma unroll
  for (int tn = 0; tn < 4; ++tn) {
#pragma unroll
    for (int r = 0; r < 8; ++r) {
      const int row = tileM + wave * 16 + lh * 8 + r;
      const int col = tileN + tn * 16 + lm;
      float v = acc[tn][r] + bias[col];
      if (GELU) v = 0.5f * v * (1.0f + erff(v * 0.70710678118654752f));
      if (RESIDUAL) v += res[(size_t)row * N + col];
      if (OUT_BF16) ((bf16*)outp)[(size_t)row * N + col] = (bf16)v;
      else          ((float*)outp)[(size_t)row * N + col] = v;
    }
  }
}

// ---------------------------------------------------------------------------
// Flash cross-attention, all-WMMA, double-buffered K/V tiles, ONE barrier
// per 32-key tile. Q fragments hoisted into registers (loaded once).
// Softmax row stats reduced with DPP16 row_ror butterflies (VALU only).
// Grid (Sq/64, H, B), 128 threads = 4 waves; wave owns 16 q rows.
// ---------------------------------------------------------------------------
__global__ __launch_bounds__(128) void attn_kernel(
    const bf16* __restrict__ q, const bf16* __restrict__ k,
    const bf16* __restrict__ v, const int* __restrict__ mask,
    bf16* __restrict__ out, int Sq, int Sk) {
  const int bb = blockIdx.z, hh = blockIdx.y;
  const int qbase = blockIdx.x * 64;
  const int t = threadIdx.x, wave = t >> 5, lane = t & 31;
  const int lm = lane & 15, lh = lane >> 4;

  __shared__ __align__(16) bf16 sQ[64][72];        // [qrow][d] (144B stride)
  __shared__ __align__(16) bf16 sK[2][32][72];     // [krow][d]
  __shared__ __align__(16) bf16 sVt[2][64][40];    // [d][k]    (80B stride)
  __shared__ __align__(16) bf16 sP[4][16][40];     // per-wave P

  {  // Q tile (once): 64x64, 32 bf16 per thread, async
    const int r = t >> 1, c = (t & 1) * 32;
    const bf16* src = q + (size_t)(bb * Sq + qbase + r) * E_DIM + hh * D_HEAD + c;
#pragma unroll
    for (int j = 0; j < 4; ++j) async_copy16B(&sQ[r][c + 8 * j], src + 8 * j);
  }

  const int kr = t >> 2, kc = (t & 3) * 16;  // K/V tile coords (16 bf16/thread)
  auto load_kv = [&](int k0, int buf) {
    const bf16* ks = k + (size_t)(bb * Sk + k0 + kr) * E_DIM + hh * D_HEAD + kc;
    async_copy16B(&sK[buf][kr][kc],     ks);
    async_copy16B(&sK[buf][kr][kc + 8], ks + 8);
    const bf16* vs = v + (size_t)(bb * Sk + k0 + kr) * E_DIM + hh * D_HEAD + kc;
#pragma unroll
    for (int j = 0; j < 16; ++j) sVt[buf][kc + j][kr] = vs[j];  // transpose
  };

  v8f acc_o[4];
#pragma unroll
  for (int i = 0; i < 4; ++i)
#pragma unroll
    for (int r = 0; r < 8; ++r) acc_o[i][r] = 0.f;
  float mrow[8], lrow[8];
#pragma unroll
  for (int r = 0; r < 8; ++r) { mrow[r] = -1e30f; lrow[r] = 0.f; }

  const int nk = Sk / 32;
  load_kv(0, 0);
  async_wait();      // Q + K/V tile 0 arrived (this wave)
  __syncthreads();   // all waves' tiles visible

  // Hoist Q fragments: sQ is loop-invariant, keep it in registers.
  const bf16* qrow = &sQ[wave * 16 + lm][0];
  const v16bf aq0 = frag_a_lds(qrow, lh);       // d 0..31
  const v16bf aq1 = frag_a_lds(qrow + 32, lh);  // d 32..63

  for (int it = 0; it < nk; ++it) {
    if (it > 0) {
      async_wait();
      __syncthreads();  // tile[buf] visible; buf^1 free to overwrite
    }
    if (it + 1 < nk) load_kv((it + 1) * 32, (it + 1) & 1);
    const int buf = it & 1;
    const int k0 = it * 32;

    // Scores: S[16 x 32] = Q(16x64) . K^T, two 16x16 tiles, 2 WMMAs each.
    v8f sc[2];
#pragma unroll
    for (int tn = 0; tn < 2; ++tn) {
      const v16bf bk0 = frag_b_lds(&sK[buf][tn * 16 + lm][0], lh);
      const v16bf bk1 = frag_b_lds(&sK[buf][tn * 16 + lm][32], lh);
      v8f s0;
#pragma unroll
      for (int r = 0; r < 8; ++r) s0[r] = 0.f;
      s0 = wmma_bf16(aq0, bk0, s0);
      s0 = wmma_bf16(aq1, bk1, s0);
      sc[tn] = s0;
    }

    // scale + padding mask (column = key index, per lane)
#pragma unroll
    for (int tn = 0; tn < 2; ++tn) {
      const int kc2 = k0 + tn * 16 + lm;
      const bool dead = (mask[(size_t)bb * Sk + kc2] == 0);
#pragma unroll
      for (int r = 0; r < 8; ++r)
        sc[tn][r] = dead ? -1e30f : sc[tn][r] * ATTN_SCALE;
    }

    // Online softmax per row; DPP row_ror butterflies keep this in the VALU.
#pragma unroll
    for (int r = 0; r < 8; ++r) {
      const float rm = row16_max(fmaxf(sc[0][r], sc[1][r]));
      const float mnew = fmaxf(mrow[r], rm);
      const float corr = expf(mrow[r] - mnew);
      const float p0 = expf(sc[0][r] - mnew);
      const float p1 = expf(sc[1][r] - mnew);
      const float ps = row16_sum(p0 + p1);
      lrow[r] = lrow[r] * corr + ps;
      mrow[r] = mnew;
#pragma unroll
      for (int tn = 0; tn < 4; ++tn) acc_o[tn][r] *= corr;
      // Re-layout P (C-frag) through per-wave LDS into A-frag order.
      sP[wave][lh * 8 + r][lm]      = (bf16)p0;
      sP[wave][lh * 8 + r][16 + lm] = (bf16)p1;
    }

    // O += P(16x32) . V(32x64): same-wave DS ordering covers sP RAW.
    const v16bf ap = frag_a_lds(&sP[wave][lm][0], lh);
#pragma unroll
    for (int tn = 0; tn < 4; ++tn) {
      const v16bf bv = frag_b_lds(&sVt[buf][tn * 16 + lm][0], lh);
      acc_o[tn] = wmma_bf16(ap, bv, acc_o[tn]);
    }
  }

  // Normalize and store (bf16, feeds out-projection GEMM).
#pragma unroll
  for (int tn = 0; tn < 4; ++tn) {
#pragma unroll
    for (int r = 0; r < 8; ++r) {
      const int row = qbase + wave * 16 + lh * 8 + r;
      const int col = hh * D_HEAD + tn * 16 + lm;
      const float o = acc_o[tn][r] / lrow[r];
      out[(size_t)(bb * Sq + row) * E_DIM + col] = (bf16)o;
    }
  }
}

// ---------------------------------------------------------------------------
// Host orchestration
// ---------------------------------------------------------------------------
extern "C" void kernel_launch(void* const* d_in, const int* in_sizes, int n_in,
                              void* d_out, int out_size, void* d_ws, size_t ws_size,
                              hipStream_t stream) {
  (void)in_sizes; (void)n_in; (void)out_size; (void)ws_size;
  const int B = 4, S1 = 2048, S2 = 512, E = E_DIM, FF = FF_DIM, H = H_DIM;
  const int T1 = B * S1, T2 = B * S2;  // 8192, 2048 tokens

  const float* x1   = (const float*)d_in[0];
  const float* x2   = (const float*)d_in[1];
  const int*   m1   = (const int*)d_in[2];
  const int*   m2   = (const int*)d_in[3];
  const float* cos1 = (const float*)d_in[4];
  const float* sin1 = (const float*)d_in[5];
  const float* cos2 = (const float*)d_in[6];
  const float* sin2 = (const float*)d_in[7];

  // params flattened in setup_inputs() insertion order, leaf order (w,b)/(g,b)
  int p = 8;
  const float *Wq1 = (const float*)d_in[p++], *Bq1 = (const float*)d_in[p++];
  const float *Wk2 = (const float*)d_in[p++], *Bk2 = (const float*)d_in[p++];
  const float *Wv2 = (const float*)d_in[p++], *Bv2 = (const float*)d_in[p++];
  const float *Wq2 = (const float*)d_in[p++], *Bq2 = (const float*)d_in[p++];
  const float *Wk1 = (const float*)d_in[p++], *Bk1 = (const float*)d_in[p++];
  const float *Wv1 = (const float*)d_in[p++], *Bv1 = (const float*)d_in[p++];
  const float *Wo1 = (const float*)d_in[p++], *Bo1 = (const float*)d_in[p++];
  const float *Wo2 = (const float*)d_in[p++], *Bo2 = (const float*)d_in[p++];
  const float *Wf1i = (const float*)d_in[p++], *Bf1i = (const float*)d_in[p++];
  const float *Wf1o = (const float*)d_in[p++], *Bf1o = (const float*)d_in[p++];
  const float *Wf2i = (const float*)d_in[p++], *Bf2i = (const float*)d_in[p++];
  const float *Wf2o = (const float*)d_in[p++], *Bf2o = (const float*)d_in[p++];
  const float *Gn11 = (const float*)d_in[p++], *Bn11 = (const float*)d_in[p++];
  const float *Gn12 = (const float*)d_in[p++], *Bn12 = (const float*)d_in[p++];
  const float *Gn21 = (const float*)d_in[p++], *Bn21 = (const float*)d_in[p++];
  const float *Gn22 = (const float*)d_in[p++], *Bn22 = (const float*)d_in[p++];
  const float *Gf1l = (const float*)d_in[p++], *Bf1l = (const float*)d_in[p++];
  const float *Gf2l = (const float*)d_in[p++], *Bf2l = (const float*)d_in[p++];

  // Workspace bump allocator (256B aligned).
  char* ws = (char*)d_ws;
  size_t off = 0;
  auto alloc = [&](size_t bytes) -> void* {
    void* r = ws + off;
    off = (off + bytes + 255) & ~(size_t)255;
    return r;
  };

  bf16* h1 = (bf16*)alloc((size_t)T1 * E * 2);  // LN(x1); reused as attn out a1
  bf16* h2 = (bf16*)alloc((size_t)T2 * E * 2);  // LN(x2); reused as attn out a2
  bf16* wq1 = (bf16*)alloc((size_t)E * E * 2);
  bf16* wk2 = (bf16*)alloc((size_t)E * E * 2);
  bf16* wv2 = (bf16*)alloc((size_t)E * E * 2);
  bf16* wq2 = (bf16*)alloc((size_t)E * E * 2);
  bf16* wk1 = (bf16*)alloc((size_t)E * E * 2);
  bf16* wv1 = (bf16*)alloc((size_t)E * E * 2);
  bf16* wo1 = (bf16*)alloc((size_t)E * E * 2);
  bf16* wo2 = (bf16*)alloc((size_t)E * E * 2);
  bf16* wf1i = (bf16*)alloc((size_t)E * FF * 2);
  bf16* wf1o = (bf16*)alloc((size_t)FF * E * 2);
  bf16* wf2i = (bf16*)alloc((size_t)E * FF * 2);
  bf16* wf2o = (bf16*)alloc((size_t)FF * E * 2);
  float* scratchF = (float*)alloc((size_t)T1 * E * 4);  // pre-rope q/k (reused)
  bf16* q1b = (bf16*)alloc((size_t)T1 * E * 2);  // reused as hn1 after attn
  bf16* k1b = (bf16*)alloc((size_t)T1 * E * 2);
  bf16* v1b = (bf16*)alloc((size_t)T1 * E * 2);
  bf16* q2b = (bf16*)alloc((size_t)T2 * E * 2);  // reused as hn2 after attn
  bf16* k2b = (bf16*)alloc((size_t)T2 * E * 2);
  bf16* v2b = (bf16*)alloc((size_t)T2 * E * 2);
  float* x1r = (float*)alloc((size_t)T1 * E * 4);
  float* x2r = (float*)alloc((size_t)T2 * E * 4);
  bf16* ff1 = (bf16*)alloc((size_t)T1 * FF * 2);  // GELU out, LN'd in place
  bf16* ff2 = (bf16*)alloc((size_t)T2 * FF * 2);

  const dim3 b256(256);
  auto conv = [&](const float* s, bf16* d, size_t n) {
    convert_f32_bf16<<<dim3(2048), b256, 0, stream>>>(s, d, n);
  };
  conv(Wq1, wq1, (size_t)E * E);  conv(Wk2, wk2, (size_t)E * E);
  conv(Wv2, wv2, (size_t)E * E);  conv(Wq2, wq2, (size_t)E * E);
  conv(Wk1, wk1, (size_t)E * E);  conv(Wv1, wv1, (size_t)E * E);
  conv(Wo1, wo1, (size_t)E * E);  conv(Wo2, wo2, (size_t)E * E);
  conv(Wf1i, wf1i, (size_t)E * FF); conv(Wf1o, wf1o, (size_t)FF * E);
  conv(Wf2i, wf2i, (size_t)E * FF); conv(Wf2o, wf2o, (size_t)FF * E);

  // 1) pre-attention LayerNorms -> bf16
  ln_kernel<float><<<dim3(T1), b256, 0, stream>>>(x1, Gn11, Bn11, h1, E);
  ln_kernel<float><<<dim3(T2), b256, 0, stream>>>(x2, Gn12, Bn12, h2, E);

  // 2) QKV projections (+ RoPE for q/k)
  const dim3 g1(E / 64, T1 / 128), g2(E / 64, T2 / 128);
  gemm_bf16_wmma<false, false, false><<<g1, b256, 0, stream>>>(h1, wq1, Bq1, nullptr, scratchF, T1, E, E);
  rope_kernel<<<dim3(T1), dim3(1024), 0, stream>>>(scratchF, cos1, sin1, q1b, S1);
  gemm_bf16_wmma<false, false, false><<<g1, b256, 0, stream>>>(h1, wk1, Bk1, nullptr, scratchF, T1, E, E);
  rope_kernel<<<dim3(T1), dim3(1024), 0, stream>>>(scratchF, cos1, sin1, k1b, S1);
  gemm_bf16_wmma<true, false, false><<<g1, b256, 0, stream>>>(h1, wv1, Bv1, nullptr, v1b, T1, E, E);
  gemm_bf16_wmma<false, false, false><<<g2, b256, 0, stream>>>(h2, wq2, Bq2, nullptr, scratchF, T2, E, E);
  rope_kernel<<<dim3(T2), dim3(1024), 0, stream>>>(scratchF, cos2, sin2, q2b, S2);
  gemm_bf16_wmma<false, false, false><<<g2, b256, 0, stream>>>(h2, wk2, Bk2, nullptr, scratchF, T2, E, E);
  rope_kernel<<<dim3(T2), dim3(1024), 0, stream>>>(scratchF, cos2, sin2, k2b, S2);
  gemm_bf16_wmma<true, false, false><<<g2, b256, 0, stream>>>(h2, wv2, Bv2, nullptr, v2b, T2, E, E);

  // 3) cross attention (a1 -> h1, a2 -> h2; LN buffers are dead now)
  attn_kernel<<<dim3(S1 / 64, H, B), dim3(128), 0, stream>>>(q1b, k2b, v2b, m2, h1, S1, S2);
  attn_kernel<<<dim3(S2 / 64, H, B), dim3(128), 0, stream>>>(q2b, k1b, v1b, m1, h2, S2, S1);

  // 4) output projections + residual -> f32 x1r/x2r
  gemm_bf16_wmma<false, false, true><<<g1, b256, 0, stream>>>(h1, wo1, Bo1, x1, x1r, T1, E, E);
  gemm_bf16_wmma<false, false, true><<<g2, b256, 0, stream>>>(h2, wo2, Bo2, x2, x2r, T2, E, E);

  // 5) FFN: LN -> (x W_in + b, GELU) -> LN -> (x W_out + b) + residual
  bf16* hn1 = q1b;  // reuse (q buffers dead after attention)
  bf16* hn2 = q2b;
  ln_kernel<float><<<dim3(T1), b256, 0, stream>>>(x1r, Gn21, Bn21, hn1, E);
  ln_kernel<float><<<dim3(T2), b256, 0, stream>>>(x2r, Gn22, Bn22, hn2, E);

  const dim3 gf1(FF / 64, T1 / 128), gf2(FF / 64, T2 / 128);
  gemm_bf16_wmma<true, true, false><<<gf1, b256, 0, stream>>>(hn1, wf1i, Bf1i, nullptr, ff1, T1, FF, E);
  gemm_bf16_wmma<true, true, false><<<gf2, b256, 0, stream>>>(hn2, wf2i, Bf2i, nullptr, ff2, T2, FF, E);

  ln_kernel<bf16><<<dim3(T1), b256, 0, stream>>>(ff1, Gf1l, Bf1l, ff1, FF);  // in place
  ln_kernel<bf16><<<dim3(T2), b256, 0, stream>>>(ff2, Gf2l, Bf2l, ff2, FF);

  float* out1p = (float*)d_out;
  float* out2p = out1p + (size_t)T1 * E;
  gemm_bf16_wmma<false, false, true><<<g1, b256, 0, stream>>>(ff1, wf1o, Bf1o, x1r, out1p, T1, E, FF);
  gemm_bf16_wmma<false, false, true><<<g2, b256, 0, stream>>>(ff2, wf2o, Bf2o, x2r, out2p, T2, E, FF);
}
